// CURAttention_34007551049919
// MI455X (gfx1250) — matrix operand order repaired and checked by reading, not compiled, not run on hardware
//
#include <hip/hip_runtime.h>
#include <hip/hip_bf16.h>

// CUR/Nystrom attention for MI455X (gfx1250, wave32, WMMA + TDM).
// B=8,H=8,N=4096,D=64,M=64. bf16 WMMA for the N-sized GEMMs, fp32 WMMA for
// the 64x64 Newton-Schulz chain, flash-style kernel_3@V, and Tensor Data
// Mover (tensor_load_to_lds + s_wait_tensorcnt) for the streamed K tiles.

#define BH   64
#define NN   4096
#define DD   64
#define MM   64

typedef __attribute__((ext_vector_type(16))) __bf16 v16bf;
typedef __attribute__((ext_vector_type(8)))  __bf16 v8bf;
typedef __attribute__((ext_vector_type(8)))  float  v8f;
typedef __attribute__((ext_vector_type(2)))  float  v2f;
typedef unsigned int u32x4 __attribute__((ext_vector_type(4)));
typedef int          i32x8 __attribute__((ext_vector_type(8)));
typedef int          i32x4 __attribute__((ext_vector_type(4)));

#if defined(__gfx1250__) && __has_builtin(__builtin_amdgcn_tensor_load_to_lds)
#define USE_TDM 1
#else
#define USE_TDM 0
#endif

// ---- WMMA fragment loads (wave32 layouts per CDNA5 ISA 7.12.2) -------------
// A (16x32 bf16): lane holds row M=lane&15; element e -> K = k0 + (e<8?0:16) + 8*hi + (e&7)
__device__ __forceinline__ v16bf ldA16(const __bf16* row, int k0, int hi) {
  v8bf lo = *reinterpret_cast<const v8bf*>(row + k0 + hi * 8);
  v8bf up = *reinterpret_cast<const v8bf*>(row + k0 + 16 + hi * 8);
  return __builtin_shufflevector(lo, up, 0,1,2,3,4,5,6,7,8,9,10,11,12,13,14,15);
}
// B (32x16 bf16): lane holds col N=lane&15; element e -> K = k0 + (e<8?0:16) + 8*hi + (e&7)
__device__ __forceinline__ v16bf ldB16(const __bf16* row, int k0, int hi) {
  v8bf lo = *reinterpret_cast<const v8bf*>(row + k0 + hi * 8);
  v8bf up = *reinterpret_cast<const v8bf*>(row + k0 + 16 + hi * 8);
  return __builtin_shufflevector(lo, up, 0,1,2,3,4,5,6,7,8,9,10,11,12,13,14,15);
}
// B fragment built from an fp32 LDS row (TDM lands raw fp32), cvt to bf16
__device__ __forceinline__ v16bf ldB16_f32(const float* row, int k0, int hi) {
  v16bf b;
  const float* p = row + k0 + hi * 8;
#pragma unroll
  for (int e = 0; e < 16; e++) b[e] = (__bf16)p[e + (e >= 8 ? 8 : 0)];
  return b;
}
__device__ __forceinline__ v8f wmma_bf(v16bf a, v16bf b, v8f c) {
  return __builtin_amdgcn_wmma_f32_16x16x32_bf16(false, a, false, b, (short)0, c, false, false);
}
// A fragment loaded w/ scale from global fp32 row, converting to bf16
__device__ __forceinline__ v16bf ldA16_f32(const float* row, int k0, int hi, float sc) {
  v16bf a;
  const float* p = row + k0 + hi * 8;
  const float* q = row + k0 + 16 + hi * 8;
#pragma unroll
  for (int e = 0; e < 8; e++) a[e] = (__bf16)(p[e] * sc);
#pragma unroll
  for (int e = 0; e < 8; e++) a[e + 8] = (__bf16)(q[e] * sc);
  return a;
}

// ---- K1: row sums + stable top-64 (argsort(-sum) semantics) ----------------
__global__ __launch_bounds__(256) void ca_topk(const float* __restrict__ Kt,
                                               const float* __restrict__ Qt,
                                               int* __restrict__ idx_c,
                                               int* __restrict__ idx_r) {
  int bh = blockIdx.x, z = blockIdx.y, tid = threadIdx.x;
  const float* T = z ? Qt : Kt;
  int* out = z ? idx_r : idx_c;
  __shared__ float s[NN];
  __shared__ float rv[256];
  __shared__ int   ri[256];
  for (int n = tid; n < NN; n += 256) {
    const float* row = T + ((size_t)bh * NN + n) * DD;
    float acc = 0.f;
    for (int d = 0; d < DD; d += 4) {
      float4 f = *(const float4*)(row + d);
      acc += f.x + f.y + f.z + f.w;
    }
    s[n] = acc;
  }
  __syncthreads();
  for (int it = 0; it < MM; it++) {
    float bv = -3.4e38f; int bi = 0x7fffffff;
    for (int n = tid; n < NN; n += 256) {
      float v = s[n];
      if (v > bv || (v == bv && n < bi)) { bv = v; bi = n; }
    }
    rv[tid] = bv; ri[tid] = bi;
    __syncthreads();
    for (int off = 128; off > 0; off >>= 1) {
      if (tid < off) {
        float v2 = rv[tid + off]; int i2 = ri[tid + off];
        if (v2 > rv[tid] || (v2 == rv[tid] && i2 < ri[tid])) { rv[tid] = v2; ri[tid] = i2; }
      }
      __syncthreads();
    }
    if (tid == 0) { out[bh * MM + it] = ri[0]; s[ri[0]] = -3.4e38f; }
    __syncthreads();
  }
}

// ---- K2: gather bf16 landmarks; init scalar ---------------------------------
__global__ __launch_bounds__(256) void ca_gather(const float* __restrict__ Kt,
                                                 const float* __restrict__ Qt,
                                                 const int* __restrict__ idx_c,
                                                 const int* __restrict__ idx_r,
                                                 __bf16* __restrict__ nc,
                                                 __bf16* __restrict__ nr,
                                                 int* __restrict__ scal) {
  int bh = blockIdx.x, tid = threadIdx.x;
  if (bh == 0 && tid == 0) *scal = 0;
  for (int e = tid; e < MM * DD; e += 256) {
    int i = e >> 6, d = e & 63;
    int ci = idx_c[bh * MM + i];
    int rr = idx_r[bh * MM + i];
    nc[(size_t)bh * MM * DD + e] = (__bf16)Kt[((size_t)bh * NN + ci) * DD + d];
    nr[(size_t)bh * MM * DD + e] = (__bf16)(Qt[((size_t)bh * NN + rr) * DD + d] * 0.125f);
  }
}

// ---- K3: kernel_1 = softmax(Qs @ nc^T), 128 rows/block ----------------------
__global__ __launch_bounds__(256) void ca_k1(const float* __restrict__ Q,
                                             const __bf16* __restrict__ nc,
                                             __bf16* __restrict__ k1) {
  int bh = blockIdx.y, n0 = blockIdx.x * 128;
  __shared__ __bf16 ncl[MM * 80];    // nc[m][d], padded row
  __shared__ float  cs[128 * 68];    // score tile
  int tid = threadIdx.x, lane = tid & 31, w = tid >> 5, li = lane & 15, hi = lane >> 4;
  for (int e = tid; e < MM * DD; e += 256)
    ncl[(e >> 6) * 80 + (e & 63)] = nc[(size_t)bh * MM * DD + e];
  __syncthreads();

  int n = n0 + w * 16 + li;
  const float* qp = Q + ((size_t)bh * NN + n) * DD;
  v8f acc[4] = {};
#pragma unroll
  for (int k0 = 0; k0 < DD; k0 += 32) {
    v16bf a = ldA16_f32(qp, k0, hi, 0.125f);
#pragma unroll
    for (int j = 0; j < 4; j++) {
      v16bf b = ldB16(ncl + (j * 16 + li) * 80, k0, hi);
      acc[j] = wmma_bf(a, b, acc[j]);
    }
  }
#pragma unroll
  for (int j = 0; j < 4; j++)
#pragma unroll
    for (int v = 0; v < 8; v++)
      cs[(w * 16 + v + 8 * hi) * 68 + j * 16 + li] = acc[j][v];
  __syncthreads();

  if (tid < 128) {
    const float* rp = cs + tid * 68;
    float mx = rp[0];
    for (int c = 1; c < MM; c++) mx = fmaxf(mx, rp[c]);
    float sum = 0.f;
    for (int c = 0; c < MM; c++) sum += __expf(rp[c] - mx);
    float is = 1.f / sum;
    __bf16* op = k1 + ((size_t)bh * NN + n0 + tid) * MM;
    for (int c = 0; c < MM; c++) op[c] = (__bf16)(__expf(rp[c] - mx) * is);
  }
}

// ---- K4: flash-style RV = softmax(nr @ K^T) @ V, one block per (b,h) --------
// K tiles arrive via the Tensor Data Mover (fp32, LDS rows padded to 68 DW by
// the TDM pad feature); V tiles staged transposed by VALU.
__global__ __launch_bounds__(256) void ca_rv(const float* __restrict__ Kt,
                                             const float* __restrict__ Vt,
                                             const __bf16* __restrict__ nr,
                                             float* __restrict__ rvws) {
  int bh = blockIdx.x;
  __shared__ __bf16 nrl[MM * 80];   // nr rows
  __shared__ float  ktf[64 * 68];   // K tile fp32 [n][d] (TDM destination)
  __shared__ __bf16 vtT[64 * 72];   // V tile transposed [d][n]
  __shared__ float  ssm[64 * 68];   // scores [m][n_local]
  __shared__ __bf16 psm[64 * 72];   // exp(scores) [m][n_local]
  __shared__ float  alpha[64];
  __shared__ float  rs[64];
  int tid = threadIdx.x, lane = tid & 31, w = tid >> 5, li = lane & 15, hi = lane >> 4;
  int rt = w & 3, ct0 = (w >> 2) * 2;

  for (int e = tid; e < MM * DD; e += 256)
    nrl[(e >> 6) * 80 + (e & 63)] = nr[(size_t)bh * MM * DD + e];
  v8f accv0 = {}, accv1 = {};
  float rmax = -3.0e38f, rsum = 0.f;   // live for tid<64 (row owners)
  __syncthreads();

  for (int nb = 0; nb < NN; nb += 64) {
#if USE_TDM
    if (w == 0) {
      // 2D TDM descriptor: 64x64 fp32 tile, tensor row stride 64,
      // LDS pad: 4 DWORDs after every 64 DWORDs -> pitch 68 floats.
      unsigned long long ga =
          (unsigned long long)(const void*)(Kt + ((size_t)bh * NN + nb) * DD);
      unsigned lds = (unsigned)(unsigned long long)(void*)ktf;
      u32x4 g0;
      g0[0] = 1u;                                   // count=1, user descriptor
      g0[1] = lds;                                  // lds_addr
      g0[2] = (unsigned)ga;                         // global_addr[31:0]
      g0[3] = (unsigned)(ga >> 32) | 0x80000000u;   // global_addr[56:32] | type=2
      i32x8 g1;
      g1[0] = (int)((2u << 16) |                    // data_size = 4 bytes
                    (1u << 20) |                    // pad_enable
                    (5u << 22) |                    // pad_interval = 64 DWORDs
                    (3u << 25));                    // pad_amount   = 4 DWORDs
      g1[1] = (int)(64u << 16);                     // tensor_dim0 = 64
      g1[2] = (int)(4096u << 16);                   // tensor_dim1[15:0] = 4096
      g1[3] = (int)(64u << 16);                     // tensor_dim1 hi | tile_dim0 = 64
      g1[4] = 64;                                   // tile_dim1 = 64
      g1[5] = 64;                                   // tensor_dim0_stride = 64
      g1[6] = 0;
      g1[7] = 0;
      i32x4 z4 = {0, 0, 0, 0};                      // groups 2/3 unused (2D)
      i32x8 z8 = {0, 0, 0, 0, 0, 0, 0, 0};
      __builtin_amdgcn_tensor_load_to_lds(g0, g1, z4, z4, z8, 0);
      __builtin_amdgcn_s_wait_tensorcnt(0);
    }
#else
    for (int i = tid; i < 64 * 16; i += 256) {
      int r = i >> 4, c4 = (i & 15) * 4;
      float4 kf = *(const float4*)(Kt + ((size_t)bh * NN + nb + r) * DD + c4);
      *(float4*)(ktf + r * 68 + c4) = kf;
    }
#endif
    // stage V tile fp32->bf16 transposed
    for (int i = tid; i < 64 * 16; i += 256) {
      int r = i >> 4, c4 = (i & 15) * 4;
      float4 vf = *(const float4*)(Vt + ((size_t)bh * NN + nb + r) * DD + c4);
      vtT[(c4 + 0) * 72 + r] = (__bf16)vf.x; vtT[(c4 + 1) * 72 + r] = (__bf16)vf.y;
      vtT[(c4 + 2) * 72 + r] = (__bf16)vf.z; vtT[(c4 + 3) * 72 + r] = (__bf16)vf.w;
    }
    // prefetch next tiles into cache while this one computes
    if (nb + 64 < NN) {
      __builtin_prefetch(Kt + ((size_t)bh * NN + nb + 64 + (tid & 63)) * DD, 0, 1);
      __builtin_prefetch(Vt + ((size_t)bh * NN + nb + 64 + (tid & 63)) * DD, 0, 1);
    }
    __syncthreads();

    // S tile = nr @ K^T  (B fragments cvt'd from the TDM-landed fp32 tile)
    v8f s0 = {}, s1 = {};
    const __bf16* ar = nrl + (rt * 16 + li) * 80;
#pragma unroll
    for (int k0 = 0; k0 < DD; k0 += 32) {
      v16bf a  = ldA16(ar, k0, hi);
      v16bf b0 = ldB16_f32(ktf + (ct0 * 16 + li) * 68, k0, hi);
      v16bf b1 = ldB16_f32(ktf + ((ct0 + 1) * 16 + li) * 68, k0, hi);
      s0 = wmma_bf(a, b0, s0);
      s1 = wmma_bf(a, b1, s1);
    }
#pragma unroll
    for (int v = 0; v < 8; v++) {
      int r = rt * 16 + v + 8 * hi;
      ssm[r * 68 + ct0 * 16 + li] = s0[v];
      ssm[r * 68 + (ct0 + 1) * 16 + li] = s1[v];
    }
    __syncthreads();

    // online softmax per landmark row (thread t owns row t)
    if (tid < 64) {
      const float* sp = ssm + tid * 68;
      float mx = sp[0];
      for (int c = 1; c < 64; c++) mx = fmaxf(mx, sp[c]);
      float nm = fmaxf(rmax, mx);
      float al = __expf(rmax - nm);
      float ps = 0.f;
      __bf16* pp = psm + tid * 72;
      for (int c = 0; c < 64; c++) {
        float p = __expf(sp[c] - nm);
        ps += p;
        pp[c] = (__bf16)p;
      }
      rsum = rsum * al + ps;
      rmax = nm;
      alpha[tid] = al;
    }
    __syncthreads();

    // rescale accumulators, then accv += P @ V
#pragma unroll
    for (int v = 0; v < 8; v++) {
      float al = alpha[rt * 16 + v + 8 * hi];
      accv0[v] *= al; accv1[v] *= al;
    }
    const __bf16* pr = psm + (rt * 16 + li) * 72;
#pragma unroll
    for (int k0 = 0; k0 < 64; k0 += 32) {
      v16bf a  = ldA16(pr, k0, hi);
      v16bf b0 = ldB16(vtT + (ct0 * 16 + li) * 72, k0, hi);
      v16bf b1 = ldB16(vtT + ((ct0 + 1) * 16 + li) * 72, k0, hi);
      accv0 = wmma_bf(a, b0, accv0);
      accv1 = wmma_bf(a, b1, accv1);
    }
    __syncthreads();
  }

  if (tid < 64) rs[tid] = rsum;
  __syncthreads();
#pragma unroll
  for (int v = 0; v < 8; v++) {
    int m = rt * 16 + v + 8 * hi;
    float inv = 1.f / rs[m];
    rvws[((size_t)bh * MM + m) * DD + ct0 * 16 + li] = accv0[v] * inv;
    rvws[((size_t)bh * MM + m) * DD + (ct0 + 1) * 16 + li] = accv1[v] * inv;
  }
}

// ---- K5: u = kernel_1[r_idx], batch-global max of column sums ---------------
__global__ __launch_bounds__(256) void ca_u(const __bf16* __restrict__ k1,
                                            const int* __restrict__ idx_r,
                                            float* __restrict__ uws,
                                            int* __restrict__ scal) {
  int bh = blockIdx.x, tid = threadIdx.x;
  __shared__ float ul[MM * MM];
  __shared__ float red[64];
  for (int e = tid; e < MM * MM; e += 256) {
    int i = e >> 6, j = e & 63;
    int r = idx_r[bh * MM + i];
    float v = (float)k1[((size_t)bh * NN + r) * MM + j];
    ul[e] = v;
    uws[(size_t)bh * MM * MM + e] = v;
  }
  __syncthreads();
  if (tid < 64) {
    float cs = 0.f;
    for (int i = 0; i < MM; i++) cs += ul[i * MM + tid];
    red[tid] = cs;
  }
  __syncthreads();
  if (tid == 0) {
    float mx = red[0];
    for (int i = 1; i < 64; i++) mx = fmaxf(mx, red[i]);
    atomicMax(scal, __float_as_int(mx));   // colsums are >= 0: int-bit max ok
  }
}

// ---- fp32 WMMA 64x64x64 matmul, 8 waves x 2 tiles, pitch 64 -----------------
__device__ __forceinline__ void mm64(float* Dm, const float* Am, const float* Bm) {
  const int tid = threadIdx.x;
  const int lane = tid & 31, w = tid >> 5;
  const int li = lane & 15, hi = lane >> 4;
  const int rt = w & 3, ct0 = (w >> 2) * 2;
  v8f a0 = {}, a1 = {};
  const float* arow = Am + (rt * 16 + li) * 64;
#pragma unroll 4
  for (int kb = 0; kb < 64; kb += 4) {
    v2f av; av[0] = arow[kb + hi * 2]; av[1] = arow[kb + hi * 2 + 1];
    const float* bp = Bm + (kb + hi * 2) * 64;
    v2f b0; b0[0] = bp[ct0 * 16 + li];       b0[1] = bp[64 + ct0 * 16 + li];
    v2f b1; b1[0] = bp[(ct0 + 1) * 16 + li]; b1[1] = bp[64 + (ct0 + 1) * 16 + li];
    a0 = __builtin_amdgcn_wmma_f32_16x16x4_f32(false, av, false, b0, (short)0, a0, false, false);
    a1 = __builtin_amdgcn_wmma_f32_16x16x4_f32(false, av, false, b1, (short)0, a1, false, false);
  }
#pragma unroll
  for (int v = 0; v < 8; v++) {
    int r = rt * 16 + v + 8 * hi;
    Dm[r * 64 + ct0 * 16 + li] = a0[v];
    Dm[r * 64 + (ct0 + 1) * 16 + li] = a1[v];
  }
}
__device__ __forceinline__ void ewdiag(float* D, const float* S, float cd) {
  for (int e = threadIdx.x; e < 4096; e += 256) {
    int i = e >> 6, j = e & 63;
    D[e] = ((i == j) ? cd : 0.f) - S[e];
  }
}

// ---- K6: Newton-Schulz pinv (6 iters) + Z^T = (Vinv @ RV)^T in bf16 ---------
__global__ __launch_bounds__(256) void ca_ns(const float* __restrict__ uws,
                                             const float* __restrict__ rvws,
                                             const int* __restrict__ scal,
                                             __bf16* __restrict__ zt) {
  int bh = blockIdx.x, tid = threadIdx.x;
  __shared__ float Vb[64 * 64];
  __shared__ float B1[64 * 64];
  __shared__ float S1[64 * 64];
  __shared__ float S2[64 * 64];
  const float* U = uws + (size_t)bh * 4096;
  float inv = 1.f / __int_as_float(*scal);
  for (int e = tid; e < 4096; e += 256) {
    int i = e >> 6, j = e & 63;
    Vb[i * 64 + j] = U[j * 64 + i] * inv;     // Vm = u^T / max(colsum)
  }
  __syncthreads();
  for (int it = 0; it < 6; it++) {
    mm64(B1, U, Vb);        __syncthreads();  // KV
    ewdiag(S1, B1, 7.f);    __syncthreads();
    mm64(S2, B1, S1);       __syncthreads();
    ewdiag(S1, S2, 15.f);   __syncthreads();
    mm64(S2, B1, S1);       __syncthreads();
    ewdiag(S1, S2, 13.f);   __syncthreads();
    mm64(S2, Vb, S1);       __syncthreads();
    for (int e = tid; e < 4096; e += 256) Vb[e] = 0.25f * S2[e];
    __syncthreads();
  }
  mm64(S1, Vb, rvws + (size_t)bh * 4096);     // Z = Vinv @ RV (B from global)
  __syncthreads();
  for (int e = tid; e < 4096; e += 256) {
    int i = e >> 6, j = e & 63;
    zt[(size_t)bh * 4096 + j * 64 + i] = (__bf16)S1[i * 64 + j];  // store Z^T
  }
}

// ---- K7: X = kernel_1 @ Z ---------------------------------------------------
__global__ __launch_bounds__(256) void ca_x(const __bf16* __restrict__ k1,
                                            const __bf16* __restrict__ zt,
                                            float* __restrict__ out) {
  int bh = blockIdx.y, n0 = blockIdx.x * 128;
  __shared__ __bf16 zl[64 * 80];   // Z^T rows: zl[d][m]
  int tid = threadIdx.x, lane = tid & 31, w = tid >> 5, li = lane & 15, hi = lane >> 4;
  for (int e = tid; e < 4096; e += 256)
    zl[(e >> 6) * 80 + (e & 63)] = zt[(size_t)bh * 4096 + e];
  __syncthreads();

  int n = n0 + w * 16 + li;
  const __bf16* ar = k1 + ((size_t)bh * NN + n) * MM;
  v8f acc[4] = {};
#pragma unroll
  for (int k0 = 0; k0 < MM; k0 += 32) {
    v16bf a = ldA16(ar, k0, hi);
#pragma unroll
    for (int j = 0; j < 4; j++) {
      v16bf b = ldB16(zl + (j * 16 + li) * 80, k0, hi);
      acc[j] = wmma_bf(a, b, acc[j]);
    }
  }
#pragma unroll
  for (int j = 0; j < 4; j++)
#pragma unroll
    for (int v = 0; v < 8; v++) {
      int r = n0 + w * 16 + v + 8 * hi;
      out[((size_t)bh * NN + r) * DD + j * 16 + li] = acc[j][v];
    }
}

extern "C" void kernel_launch(void* const* d_in, const int* in_sizes, int n_in,
                              void* d_out, int out_size, void* d_ws, size_t ws_size,
                              hipStream_t stream) {
  const float* Q  = (const float*)d_in[0];
  const float* Kt = (const float*)d_in[1];
  const float* Vt = (const float*)d_in[2];
  // d_in[3] = mask, all-true for this shape: softmax masking is a no-op.
  float* X = (float*)d_out;

  char* ws = (char*)d_ws;
  size_t off = 0;
  int*    idx_c = (int*)(ws + off);    off += (size_t)BH * MM * 4;
  int*    idx_r = (int*)(ws + off);    off += (size_t)BH * MM * 4;
  __bf16* nc    = (__bf16*)(ws + off); off += (size_t)BH * MM * DD * 2;
  __bf16* nr    = (__bf16*)(ws + off); off += (size_t)BH * MM * DD * 2;
  __bf16* k1    = (__bf16*)(ws + off); off += (size_t)BH * NN * MM * 2;  // 33.5 MB
  float*  uws   = (float*)(ws + off);  off += (size_t)BH * MM * MM * 4;
  float*  rvws  = (float*)(ws + off);  off += (size_t)BH * MM * DD * 4;
  __bf16* zt    = (__bf16*)(ws + off); off += (size_t)BH * MM * DD * 2;
  int*    scal  = (int*)(ws + off);    off += 4;

  ca_topk  <<<dim3(BH, 2), 256, 0, stream>>>(Kt, Q, idx_c, idx_r);
  ca_gather<<<BH, 256, 0, stream>>>(Kt, Q, idx_c, idx_r, nc, nr, scal);
  ca_k1    <<<dim3(NN / 128, BH), 256, 0, stream>>>(Q, nc, k1);
  ca_rv    <<<BH, 256, 0, stream>>>(Kt, Vt, nr, rvws);
  ca_u     <<<BH, 256, 0, stream>>>(k1, idx_r, uws, scal);
  ca_ns    <<<BH, 256, 0, stream>>>(uws, rvws, scal, zt);
  ca_x     <<<dim3(NN / 128, BH), 256, 0, stream>>>(k1, zt, X);
}
